// SchNet_4372276707779
// MI455X (gfx1250) — matrix operand order repaired and checked by reading, not compiled, not run on hardware
//
#include <hip/hip_runtime.h>
#include <hip/hip_bf16.h>

typedef __attribute__((ext_vector_type(16))) _Float16 v16h;
typedef __attribute__((ext_vector_type(8)))  float    v8f;

#define BATCH 8
#define NA    600
#define NNBH  64
#define ROWS  (BATCH * NA)          // 4800 atom rows
#define PAIRS (ROWS * NNBH)         // 307200 (atom, neighbor) pairs
#define KPAD  32                    // 25 gaussians padded to 32 for K=32 WMMA

// per-interaction f16 weight block layout (element offsets)
#define WOFF_IN2F 0                 // [128][128]
#define WOFF_F1   16384             // [32][128] (zero padded 25->32)
#define WOFF_F2   20480             // [128][128]
#define WOFF_OW   36864             // [128][128]
#define WOFF_DW   53248             // [128][128]
#define WBLK      69632             // halves per interaction

// ---------------------------------------------------------------------------
// CDNA5 async global -> LDS copy (ASYNCcnt path) + WMMA helpers
// ---------------------------------------------------------------------------
__device__ __forceinline__ void async_copy_b128(void* lds, const void* gptr) {
  unsigned ldsOff = (unsigned)(size_t)lds;   // low 32 bits of generic = LDS offset
  asm volatile("global_load_async_to_lds_b128 %0, %1, off"
               :: "v"(ldsOff), "v"(gptr) : "memory");
}
__device__ __forceinline__ void wait_async() {
  asm volatile("s_wait_asynccnt 0x0" ::: "memory");
}

__device__ __forceinline__ v8f wmma_f16(v16h a, v16h b, v8f c) {
  return __builtin_amdgcn_wmma_f32_16x16x32_f16(
      /*neg_a=*/false, a, /*neg_b=*/false, b,
      /*c_mod=*/(short)0, c, /*reuse_a=*/false, /*reuse_b=*/false);
}

// A: 16x32 f16 tile, row-major with row stride lda (elements).
__device__ __forceinline__ v16h load_a_frag(const _Float16* A, int lda, int lane) {
  int m  = lane & 15;
  int hi = (lane >> 4) & 1;
  const _Float16* row = A + m * lda;
  v16h f;
#pragma unroll
  for (int v = 0; v < 8; ++v) {
    int kbase = (v < 4) ? (2 * v) : (16 + 2 * (v - 4));
    int k = kbase + hi * 8;
    f[2 * v]     = row[k];
    f[2 * v + 1] = row[k + 1];
  }
  return f;
}

// B: 32x16 f16 tile (KxN), row-major with row stride ldb.
__device__ __forceinline__ v16h load_b_frag(const _Float16* B, int ldb, int lane) {
  int n  = lane & 15;
  int hi = (lane >> 4) & 1;
  v16h f;
#pragma unroll
  for (int v = 0; v < 8; ++v) {
    int k = ((v < 4) ? (2 * v) : (16 + 2 * (v - 4))) + hi * 8;
    f[2 * v]     = B[k * ldb + n];
    f[2 * v + 1] = B[(k + 1) * ldb + n];
  }
  return f;
}

// shifted softplus: ln(1+e^x) - ln2, numerically stable
__device__ __forceinline__ float sspf(float x) {
  float ax = fabsf(x);
  return fmaxf(x, 0.0f) + __logf(1.0f + __expf(-ax)) - 0.69314718056f;
}

// ---------------------------------------------------------------------------
// 0. One-shot f32 -> f16 weight conversion into workspace (per interaction)
// ---------------------------------------------------------------------------
__global__ __launch_bounds__(256) void cvt_weights_kernel(
    const float* __restrict__ in2fW, const float* __restrict__ f1W,
    const float* __restrict__ f2W, const float* __restrict__ oW,
    const float* __restrict__ dW, _Float16* __restrict__ out) {
  int i = blockIdx.x * 256 + threadIdx.x;
  if (i >= WBLK) return;
  float v;
  if (i < WOFF_F1) {
    v = in2fW[i];
  } else if (i < WOFF_F2) {
    int k = (i - WOFF_F1) >> 7, n = i & 127;
    v = (k < 25) ? f1W[k * 128 + n] : 0.0f;
  } else if (i < WOFF_OW) {
    v = f2W[i - WOFF_F2];
  } else if (i < WOFF_DW) {
    v = oW[i - WOFF_OW];
  } else {
    v = dW[i - WOFF_DW];
  }
  out[i] = (_Float16)v;
}

// ---------------------------------------------------------------------------
// 1. Embedding + props dense -> x [ROWS,128]
// ---------------------------------------------------------------------------
__global__ __launch_bounds__(256) void embed_kernel(
    const int* __restrict__ Z, const float* __restrict__ props,
    const float* __restrict__ emb, const float* __restrict__ pW,
    const float* __restrict__ pb, float* __restrict__ x) {
  int idx = blockIdx.x * 256 + threadIdx.x;    // over ROWS*128
  int c = idx & 127;
  int atom = idx >> 7;
  float val;
  if (c < 64) {
    const float* pr = props + atom * 8;
    float acc = pb[c];
#pragma unroll
    for (int i = 0; i < 8; ++i) acc += pr[i] * pW[i * 64 + c];
    val = acc;
  } else {
    int z = Z[atom];
    val = emb[z * 64 + (c - 64)];
  }
  x[idx] = val;
}

// ---------------------------------------------------------------------------
// 2. Distances + gaussian smearing (padded K=32, f16) + cutoff*mask scale
// ---------------------------------------------------------------------------
__global__ __launch_bounds__(256) void smear_kernel(
    const float* __restrict__ pos, const int* __restrict__ nbrs,
    const float* __restrict__ mask, _Float16* __restrict__ fij,
    float* __restrict__ scl) {
  int idx = blockIdx.x * 256 + threadIdx.x;    // pair index
  int an = idx / NNBH;                          // b*NA + a
  int b  = an / NA;
  int j  = nbrs[idx];
  const float* pi = pos + (long)an * 3;
  const float* pj = pos + (long)(b * NA + j) * 3;
  float dx = pj[0] - pi[0], dy = pj[1] - pi[1], dz = pj[2] - pi[2];
  float d2 = dx * dx + dy * dy + dz * dz;
  float m = mask[idx];
  float r = (m > 0.0f) ? __fsqrt_rn(d2) : 0.0f;
  scl[idx] = (r <= 5.0f) ? m : 0.0f;
  const float width = 5.0f / 24.0f;
  const float coeff = -0.5f * (24.0f / 5.0f) * (24.0f / 5.0f);
  _Float16* out = fij + (long)idx * KPAD;
#pragma unroll
  for (int g = 0; g < KPAD; ++g) {
    float off = width * (float)g;
    float d = r - off;
    float v = (g < 25) ? __expf(coeff * d * d) : 0.0f;
    out[g] = (_Float16)v;
  }
}

// ---------------------------------------------------------------------------
// 3. y = x @ in2f_W  (no bias) — WMMA, 64-row tiles; weights staged async
// ---------------------------------------------------------------------------
__global__ __launch_bounds__(256) void in2f_kernel(
    const float* __restrict__ x, const _Float16* __restrict__ Wh,
    float* __restrict__ y) {
  __shared__ _Float16 sA[64][128];
  __shared__ _Float16 sB[128][128];
  int t = threadIdx.x;
  long row0 = (long)blockIdx.x * 64;
#pragma unroll
  for (int i = t; i < 2048; i += 256)               // 2048 x b128 = 32 KB
    async_copy_b128(&sB[0][0] + i * 8, Wh + i * 8);
  for (int i = t; i < 64 * 128; i += 256)
    sA[i >> 7][i & 127] = (_Float16)x[row0 * 128 + i];
  wait_async();
  __syncthreads();
  int wave = t >> 5, lane = t & 31;
  int n0 = wave * 16;
  int nn = n0 + (lane & 15);
  int hi = lane >> 4;
  for (int mt = 0; mt < 4; ++mt) {
    v8f c;
#pragma unroll
    for (int r = 0; r < 8; ++r) c[r] = 0.0f;
#pragma unroll
    for (int kt = 0; kt < 4; ++kt) {
      v16h a = load_a_frag(&sA[mt * 16][kt * 32], 128, lane);
      v16h b = load_b_frag(&sB[kt * 32][n0], 128, lane);
      c = wmma_f16(a, b, c);
    }
#pragma unroll
    for (int r = 0; r < 8; ++r)
      y[(row0 + mt * 16 + hi * 8 + r) * 128 + nn] = c[r];
  }
}

// ---------------------------------------------------------------------------
// 4. Fused filter network + CFConv aggregation. One block per atom.
//    All f16 operands (f1Wp, f2W, fij tile) staged via async LDS copies.
// ---------------------------------------------------------------------------
__global__ __launch_bounds__(256) void cfconv_kernel(
    const _Float16* __restrict__ fij, const float* __restrict__ scl,
    const int* __restrict__ nbrs, const float* __restrict__ y,
    const _Float16* __restrict__ f1h, const _Float16* __restrict__ f2h,
    const float* __restrict__ f1b, const float* __restrict__ f2b,
    float* __restrict__ agg) {
  __shared__ _Float16 sF1[32][128];    // 8 KB  (25->32 zero-padded)
  __shared__ _Float16 sF2[128][128];   // 32 KB
  __shared__ _Float16 sA[64][32];      // 4 KB  f_ij tile
  __shared__ _Float16 sH[64][128];     // 16 KB
  __shared__ float sB1[128];
  __shared__ float sB2[128];
  __shared__ float sScale[64];
  __shared__ int   sNbr[64];

  int t = threadIdx.x;
  int atomIdx = blockIdx.x;                 // 0..ROWS-1
  int bBase = (atomIdx / NA) * NA;          // batch row base for y gather
  long pair0 = (long)atomIdx * NNBH;

  // async staging: f1Wp 512 x b128, f2W 2048 x b128, fij tile 256 x b128
#pragma unroll
  for (int i = t; i < 512; i += 256)
    async_copy_b128(&sF1[0][0] + i * 8, f1h + i * 8);
#pragma unroll
  for (int i = t; i < 2048; i += 256)
    async_copy_b128(&sF2[0][0] + i * 8, f2h + i * 8);
  async_copy_b128(&sA[0][0] + t * 8, fij + pair0 * KPAD + t * 8);

  if (t < 128) { sB1[t] = f1b[t]; sB2[t] = f2b[t]; }
  if (t < 64)  { sScale[t] = scl[pair0 + t]; sNbr[t] = nbrs[pair0 + t]; }
  wait_async();
  __syncthreads();

  int wave = t >> 5, lane = t & 31;
  int n0 = wave * 16;
  int nn = n0 + (lane & 15);
  int hi = lane >> 4;

  // GEMM1 (K=32) + ssp -> sH
  for (int mt = 0; mt < 4; ++mt) {
    v8f c;
#pragma unroll
    for (int r = 0; r < 8; ++r) c[r] = sB1[nn];
    v16h a = load_a_frag(&sA[mt * 16][0], 32, lane);
    v16h b = load_b_frag(&sF1[0][n0], 128, lane);
    c = wmma_f16(a, b, c);
#pragma unroll
    for (int r = 0; r < 8; ++r)
      sH[mt * 16 + hi * 8 + r][nn] = (_Float16)sspf(c[r]);
  }
  __syncthreads();

  // GEMM2 (K=128) + scale*y gather + reduce over 64 neighbors
  float regAgg = 0.0f;
  for (int mt = 0; mt < 4; ++mt) {
    v8f c;
#pragma unroll
    for (int r = 0; r < 8; ++r) c[r] = sB2[nn];
#pragma unroll
    for (int kt = 0; kt < 4; ++kt) {
      v16h a = load_a_frag(&sH[mt * 16][kt * 32], 128, lane);
      v16h b = load_b_frag(&sF2[kt * 32][n0], 128, lane);
      c = wmma_f16(a, b, c);
    }
    float part = 0.0f;
#pragma unroll
    for (int r = 0; r < 8; ++r) {
      int row = mt * 16 + hi * 8 + r;     // neighbor slot
      int j = sNbr[row];
      float yv = y[(long)(bBase + j) * 128 + nn];
      part += c[r] * sScale[row] * yv;
    }
    part += __shfl_xor(part, 16, 32);     // merge lane halves (m 0-7 / 8-15)
    regAgg += part;
  }
  if (hi == 0) agg[(long)atomIdx * 128 + nn] = regAgg;
}

// ---------------------------------------------------------------------------
// 5. v = ssp(agg @ oW + ob) @ dW + db ;  x += v   (64-row tiles)
// ---------------------------------------------------------------------------
__global__ __launch_bounds__(256) void out_dense_kernel(
    const float* __restrict__ agg,
    const _Float16* __restrict__ oWh, const float* __restrict__ ob,
    const _Float16* __restrict__ dWh, const float* __restrict__ db,
    float* __restrict__ x) {
  __shared__ _Float16 sA[64][128];     // 16 KB (agg, then H)
  __shared__ _Float16 sW[128][128];    // 32 KB (oW, then dW)
  __shared__ float sBias[128];
  int t = threadIdx.x;
  long row0 = (long)blockIdx.x * 64;
#pragma unroll
  for (int i = t; i < 2048; i += 256)
    async_copy_b128(&sW[0][0] + i * 8, oWh + i * 8);
  for (int i = t; i < 64 * 128; i += 256)
    sA[i >> 7][i & 127] = (_Float16)agg[row0 * 128 + i];
  if (t < 128) sBias[t] = ob[t];
  wait_async();
  __syncthreads();

  int wave = t >> 5, lane = t & 31;
  int n0 = wave * 16;
  int nn = n0 + (lane & 15);
  int hi = lane >> 4;

  float h[4][8];
  for (int mt = 0; mt < 4; ++mt) {
    v8f c;
#pragma unroll
    for (int r = 0; r < 8; ++r) c[r] = sBias[nn];
#pragma unroll
    for (int kt = 0; kt < 4; ++kt) {
      v16h a = load_a_frag(&sA[mt * 16][kt * 32], 128, lane);
      v16h b = load_b_frag(&sW[kt * 32][n0], 128, lane);
      c = wmma_f16(a, b, c);
    }
#pragma unroll
    for (int r = 0; r < 8; ++r) h[mt][r] = sspf(c[r]);
  }
  __syncthreads();
  // restage: sA <- ssp result (f16), sW <- dW (async), sBias <- db
#pragma unroll
  for (int i = t; i < 2048; i += 256)
    async_copy_b128(&sW[0][0] + i * 8, dWh + i * 8);
  for (int mt = 0; mt < 4; ++mt)
#pragma unroll
    for (int r = 0; r < 8; ++r)
      sA[mt * 16 + hi * 8 + r][nn] = (_Float16)h[mt][r];
  if (t < 128) sBias[t] = db[t];
  wait_async();
  __syncthreads();

  for (int mt = 0; mt < 4; ++mt) {
    v8f c;
#pragma unroll
    for (int r = 0; r < 8; ++r) c[r] = sBias[nn];
#pragma unroll
    for (int kt = 0; kt < 4; ++kt) {
      v16h a = load_a_frag(&sA[mt * 16][kt * 32], 128, lane);
      v16h b = load_b_frag(&sW[kt * 32][n0], 128, lane);
      c = wmma_f16(a, b, c);
    }
#pragma unroll
    for (int r = 0; r < 8; ++r) {
      long row = row0 + mt * 16 + hi * 8 + r;
      x[row * 128 + nn] += c[r];          // residual update
    }
  }
}

// ---------------------------------------------------------------------------
extern "C" void kernel_launch(void* const* d_in, const int* in_sizes, int n_in,
                              void* d_out, int out_size, void* d_ws, size_t ws_size,
                              hipStream_t stream) {
  (void)in_sizes; (void)n_in; (void)out_size; (void)ws_size;
  const int*   Z     = (const int*)d_in[0];
  const float* props = (const float*)d_in[1];
  const float* pos   = (const float*)d_in[2];
  const int*   nbrs  = (const int*)d_in[3];
  const float* mask  = (const float*)d_in[4];
  const float* emb   = (const float*)d_in[5];
  const float* pW    = (const float*)d_in[6];
  const float* pb    = (const float*)d_in[7];

  float* x = (float*)d_out;                       // x lives in d_out

  char* ws = (char*)d_ws;
  size_t off = 0;
  _Float16* fij = (_Float16*)(ws + off); off += (size_t)PAIRS * KPAD * 2;  // 19,660,800
  float* scl = (float*)(ws + off);       off += (size_t)PAIRS * 4;          //  1,228,800
  float* yb  = (float*)(ws + off);       off += (size_t)ROWS * 128 * 4;     //  2,457,600
  float* agg = (float*)(ws + off);       off += (size_t)ROWS * 128 * 4;     //  2,457,600
  _Float16* wts = (_Float16*)(ws + off);  // 3 x WBLK halves = 417,792 B

  // weight pre-conversion (f32 -> f16, padded), once per interaction
  for (int it = 0; it < 3; ++it) {
    const float* in2fW = (const float*)d_in[8 + it * 9 + 0];
    const float* f1W   = (const float*)d_in[8 + it * 9 + 1];
    const float* f2W   = (const float*)d_in[8 + it * 9 + 3];
    const float* oW    = (const float*)d_in[8 + it * 9 + 5];
    const float* dW    = (const float*)d_in[8 + it * 9 + 7];
    cvt_weights_kernel<<<(WBLK + 255) / 256, 256, 0, stream>>>(
        in2fW, f1W, f2W, oW, dW, wts + (size_t)it * WBLK);
  }

  embed_kernel<<<(ROWS * 128) / 256, 256, 0, stream>>>(Z, props, emb, pW, pb, x);
  smear_kernel<<<PAIRS / 256, 256, 0, stream>>>(pos, nbrs, mask, fij, scl);

  for (int it = 0; it < 3; ++it) {
    const _Float16* w = wts + (size_t)it * WBLK;
    const float* f1b = (const float*)d_in[8 + it * 9 + 2];
    const float* f2b = (const float*)d_in[8 + it * 9 + 4];
    const float* ob  = (const float*)d_in[8 + it * 9 + 6];
    const float* db  = (const float*)d_in[8 + it * 9 + 8];

    in2f_kernel<<<ROWS / 64, 256, 0, stream>>>(x, w + WOFF_IN2F, yb);
    cfconv_kernel<<<ROWS, 256, 0, stream>>>(fij, scl, nbrs, yb,
                                            w + WOFF_F1, w + WOFF_F2,
                                            f1b, f2b, agg);
    out_dense_kernel<<<ROWS / 64, 256, 0, stream>>>(agg, w + WOFF_OW, ob,
                                                    w + WOFF_DW, db, x);
  }
}
